// SimpleGNN_69741678953163
// MI455X (gfx1250) — matrix operand order repaired and checked by reading, not compiled, not run on hardware
//
#include <hip/hip_runtime.h>
#include <hip/hip_fp16.h>

// ---------------------------------------------------------------------------
// CDNA5 (gfx1250) GCN pipeline.
// Dense GEMMs via V_WMMA_F32_16X16X32_F16, wave32, 2x2 register blocking:
// each wave computes a 32x32 output tile with 4 f32 accumulators, reusing
// each A/B fragment twice (2 b128 loads per WMMA instead of 4).
// Scatter phase via global_atomic_add_f32 (L2-resident gathers, 192MB L2).
// ---------------------------------------------------------------------------

typedef __attribute__((ext_vector_type(16))) _Float16 v16h;
typedef __attribute__((ext_vector_type(8)))  float    v8f;

union Frag16 {
    v16h  h;
    uint4 q[2];   // 32 bytes = 16 halfs
};

// ---------------------------- utility kernels ------------------------------

__global__ void k_fill_f32(float* __restrict__ p, float v, long long n) {
    long long i = (long long)blockIdx.x * blockDim.x + threadIdx.x;
    if (i < n) p[i] = v;
}

__global__ void k_deg_edges(const int* __restrict__ dst, float* __restrict__ deg, int E) {
    int i = blockIdx.x * blockDim.x + threadIdx.x;
    if (i < E) atomicAdd(&deg[dst[i]], 1.0f);
}

__global__ void k_rsqrt(const float* __restrict__ deg, float* __restrict__ dinv, int n) {
    int i = blockIdx.x * blockDim.x + threadIdx.x;
    if (i < n) dinv[i] = rsqrtf(deg[i]);
}

__global__ void k_f32_to_f16(const float* __restrict__ in, _Float16* __restrict__ out, long long n) {
    long long i = (long long)blockIdx.x * blockDim.x + threadIdx.x;
    if (i < n) out[i] = (_Float16)in[i];
}

// W is (K x Nout) row-major f32  ->  Wt is (Nout x K) row-major f16
__global__ void k_transpose_w16(const float* __restrict__ W, _Float16* __restrict__ Wt,
                                int K, int Nout) {
    int i = blockIdx.x * blockDim.x + threadIdx.x;
    if (i >= K * Nout) return;
    int k = i / Nout, n = i % Nout;
    Wt[(size_t)n * K + k] = (_Float16)W[i];
}

// ------------------------------ WMMA GEMM ----------------------------------
// Y[M x Nout] = A[M x K](f16, row-major) * Bt[Nout x K](f16, row-major)^T
// One wave per 32x32 output tile: 2x2 grid of 16x16 WMMA tiles.
// A fragment (16x32, ISA 7.12.2): lane L holds row M=L%16; K-pairs split
//   0..7 / 16..23 (lanes 0-15) and 8..15 / 24..31 (lanes 16-31).
// B fragment (32x16): lane L holds column N=L%16; same K-pair split as A:
//   0..7 / 16..23 (lanes 0-15) and 8..15 / 24..31 (lanes 16-31).
__global__ void k_wmma_gemm_f16(const _Float16* __restrict__ A,
                                const _Float16* __restrict__ Bt,
                                float* __restrict__ Y,
                                int M, int K, int Nout) {
    const int lane = threadIdx.x;               // 0..31
    const int m0 = blockIdx.x * 32;
    const int n0 = blockIdx.y * 32;

    int arow0 = m0 + (lane & 15);
    int arow1 = arow0 + 16;
    if (arow0 >= M) arow0 = M - 1;              // clamp reads; stores guarded
    if (arow1 >= M) arow1 = M - 1;
    const int koffA = (lane < 16) ? 0 : 8;      // A: split K-pair striping
    const int koffB = (lane < 16) ? 0 : 8;      // B: split K-pair striping

    const _Float16* aptr0 = A  + (size_t)arow0 * K + koffA;
    const _Float16* aptr1 = A  + (size_t)arow1 * K + koffA;
    const _Float16* bptr0 = Bt + (size_t)(n0      + (lane & 15)) * K + koffB;
    const _Float16* bptr1 = Bt + (size_t)(n0 + 16 + (lane & 15)) * K + koffB;

    v8f c00 = {}, c01 = {}, c10 = {}, c11 = {};
    for (int k0 = 0; k0 < K; k0 += 32) {
        Frag16 a0, a1, b0, b1;
        a0.q[0] = *reinterpret_cast<const uint4*>(aptr0 + k0);        // K+0..7
        a0.q[1] = *reinterpret_cast<const uint4*>(aptr0 + k0 + 16);   // K+16..23
        a1.q[0] = *reinterpret_cast<const uint4*>(aptr1 + k0);
        a1.q[1] = *reinterpret_cast<const uint4*>(aptr1 + k0 + 16);
        b0.q[0] = *reinterpret_cast<const uint4*>(bptr0 + k0);        // K+0..7
        b0.q[1] = *reinterpret_cast<const uint4*>(bptr0 + k0 + 16);   // K+16..23
        b1.q[0] = *reinterpret_cast<const uint4*>(bptr1 + k0);
        b1.q[1] = *reinterpret_cast<const uint4*>(bptr1 + k0 + 16);
        c00 = __builtin_amdgcn_wmma_f32_16x16x32_f16(false, a0.h, false, b0.h,
                                                     (short)0, c00, false, false);
        c01 = __builtin_amdgcn_wmma_f32_16x16x32_f16(false, a0.h, false, b1.h,
                                                     (short)0, c01, false, false);
        c10 = __builtin_amdgcn_wmma_f32_16x16x32_f16(false, a1.h, false, b0.h,
                                                     (short)0, c10, false, false);
        c11 = __builtin_amdgcn_wmma_f32_16x16x32_f16(false, a1.h, false, b1.h,
                                                     (short)0, c11, false, false);
    }

    // f32 C/D layout: lane holds column (lane&15); VGPR r = row r (+8 hi half)
    const int colA  = n0 + (lane & 15);
    const int colB  = colA + 16;
    const int rbase = ((lane >> 4) << 3);
#pragma unroll
    for (int r = 0; r < 8; ++r) {
        int row0 = m0 + rbase + r;
        int row1 = row0 + 16;
        if (row0 < M) {
            Y[(size_t)row0 * Nout + colA] = c00[r];
            Y[(size_t)row0 * Nout + colB] = c01[r];
        }
        if (row1 < M) {
            Y[(size_t)row1 * Nout + colA] = c10[r];
            Y[(size_t)row1 * Nout + colB] = c11[r];
        }
    }
}

// --------------------------- message passing -------------------------------

// self-loop contribution: acc[i][c] = dinv[i]^2 * Y[i][c]  (full overwrite)
__global__ void k_self_init(const float* __restrict__ Y, const float* __restrict__ dinv,
                            float* __restrict__ acc, long long total, int H) {
    long long idx = (long long)blockIdx.x * blockDim.x + threadIdx.x;
    if (idx >= total) return;
    int i = (int)(idx / H);
    float w = dinv[i];
    acc[idx] = w * w * Y[idx];
}

// per-edge: acc[dst] += dinv[src]*dinv[dst] * Y[src]   (4 channels / thread)
__global__ void k_scatter_edges(const float* __restrict__ Y, const float* __restrict__ dinv,
                                const int* __restrict__ src, const int* __restrict__ dst,
                                float* __restrict__ acc, int E, int H) {
    long long idx = (long long)blockIdx.x * blockDim.x + threadIdx.x;
    const int chunks = H >> 2;
    long long total = (long long)E * chunks;
    if (idx >= total) return;
    int e  = (int)(idx / chunks);
    int c4 = (int)(idx % chunks) << 2;
    int s = src[e], d = dst[e];
    float w = dinv[s] * dinv[d];
    const float4 y = *reinterpret_cast<const float4*>(Y + (size_t)s * H + c4);
    float* ap = acc + (size_t)d * H + c4;
    atomicAdd(ap + 0, w * y.x);
    atomicAdd(ap + 1, w * y.y);
    atomicAdd(ap + 2, w * y.z);
    atomicAdd(ap + 3, w * y.w);
}

// h_next = relu(acc + bias), emitted as f16 for the next WMMA GEMM
__global__ void k_bias_relu_f16(const float* __restrict__ acc, const float* __restrict__ bias,
                                _Float16* __restrict__ h, long long total, int H) {
    long long idx = (long long)blockIdx.x * blockDim.x + threadIdx.x;
    if (idx >= total) return;
    int c = (int)(idx % H);
    float v = acc[idx] + bias[c];
    h[idx] = (_Float16)fmaxf(v, 0.0f);
}

// ------------------------------- readout -----------------------------------

__global__ void k_pool_count(const int* __restrict__ batch, float* __restrict__ counts, int N) {
    int i = blockIdx.x * blockDim.x + threadIdx.x;
    if (i < N) atomicAdd(&counts[batch[i]], 1.0f);
}

__global__ void k_pool_accum(const _Float16* __restrict__ h, const int* __restrict__ batch,
                             float* __restrict__ pooled, long long total, int H) {
    long long idx = (long long)blockIdx.x * blockDim.x + threadIdx.x;
    if (idx >= total) return;
    int i = (int)(idx / H);
    int c = (int)(idx % H);
    atomicAdd(&pooled[(size_t)batch[i] * H + c], (float)h[idx]);
}

__global__ void k_readout(const float* __restrict__ pooled, const float* __restrict__ counts,
                          const float* __restrict__ Wout, const float* __restrict__ bout,
                          float* __restrict__ out, int G, int H, int T) {
    int idx = blockIdx.x * blockDim.x + threadIdx.x;
    if (idx >= G * T) return;
    int g = idx / T, t = idx % T;
    float cnt = fmaxf(counts[g], 1.0f);
    float sum = 0.0f;
    for (int c = 0; c < H; ++c)
        sum += pooled[(size_t)g * H + c] * Wout[c * T + t];
    out[idx] = sum / cnt + bout[t];
}

// ------------------------------- launcher ----------------------------------

extern "C" void kernel_launch(void* const* d_in, const int* in_sizes, int n_in,
                              void* d_out, int out_size, void* d_ws, size_t ws_size,
                              hipStream_t stream) {
    const float* x     = (const float*)d_in[0];
    const int*   ei    = (const int*)  d_in[1];
    const int*   batch = (const int*)  d_in[2];
    const float* W[4]  = {(const float*)d_in[3], (const float*)d_in[5],
                          (const float*)d_in[7], (const float*)d_in[9]};
    const float* b[4]  = {(const float*)d_in[4], (const float*)d_in[6],
                          (const float*)d_in[8], (const float*)d_in[10]};
    const float* Wout  = (const float*)d_in[11];
    const float* bout  = (const float*)d_in[12];
    float*       out   = (float*)d_out;

    const int N = in_sizes[2];
    const int E = in_sizes[1] / 2;
    const int F = in_sizes[0] / N;
    const int H = in_sizes[4];
    const int T = in_sizes[12];
    const int G = out_size / T;

    const int* srcIdx = ei;
    const int* dstIdx = ei + E;

    // ---- carve workspace (256B aligned slabs) ----
    char* p = (char*)d_ws;
    auto carve = [&](size_t bytes) -> char* {
        char* r = p;
        p += (bytes + 255) & ~(size_t)255;
        return r;
    };
    float*    deg    = (float*)   carve((size_t)N * 4);
    float*    dinv   = (float*)   carve((size_t)N * 4);
    _Float16* h16a   = (_Float16*)carve((size_t)N * H * 2);
    _Float16* h16b   = (_Float16*)carve((size_t)N * H * 2);
    _Float16* Wt16   = (_Float16*)carve((size_t)H * H * 2);
    float*    Ybuf   = (float*)   carve((size_t)N * H * 4);
    float*    accbuf = (float*)   carve((size_t)N * H * 4);
    float*    pooled = (float*)   carve((size_t)G * H * 4);
    float*    counts = (float*)   carve((size_t)G * 4);

    const int BT = 256;
    auto blocks = [&](long long n) { return (unsigned)((n + BT - 1) / BT); };

    // ---- degree / normalization ----
    k_fill_f32 <<<blocks(N), BT, 0, stream>>>(deg, 1.0f, N);       // self-loops
    k_deg_edges<<<blocks(E), BT, 0, stream>>>(dstIdx, deg, E);
    k_rsqrt    <<<blocks(N), BT, 0, stream>>>(deg, dinv, N);

    // ---- input features -> f16 ----
    k_f32_to_f16<<<blocks((long long)N * F), BT, 0, stream>>>(x, h16a, (long long)N * F);

    // ---- 4 GCN layers ----
    const long long NH = (long long)N * H;
    _Float16* hin  = h16a;
    _Float16* hout = h16b;
    for (int l = 0; l < 4; ++l) {
        const int K = (l == 0) ? F : H;
        k_transpose_w16<<<blocks((long long)K * H), BT, 0, stream>>>(W[l], Wt16, K, H);

        dim3 ggrid((N + 31) / 32, H / 32);      // one wave per 32x32 tile
        k_wmma_gemm_f16<<<ggrid, 32, 0, stream>>>(hin, Wt16, Ybuf, N, K, H);

        k_self_init    <<<blocks(NH), BT, 0, stream>>>(Ybuf, dinv, accbuf, NH, H);
        k_scatter_edges<<<blocks((long long)E * (H >> 2)), BT, 0, stream>>>(
            Ybuf, dinv, srcIdx, dstIdx, accbuf, E, H);
        k_bias_relu_f16<<<blocks(NH), BT, 0, stream>>>(accbuf, b[l], hout, NH, H);

        _Float16* t = hin; hin = hout; hout = t;
    }

    // ---- global mean pool + readout ----
    k_fill_f32  <<<blocks(G),                BT, 0, stream>>>(counts, 0.0f, G);
    k_fill_f32  <<<blocks((long long)G * H), BT, 0, stream>>>(pooled, 0.0f, (long long)G * H);
    k_pool_count<<<blocks(N),                BT, 0, stream>>>(batch, counts, N);
    k_pool_accum<<<blocks(NH),               BT, 0, stream>>>(hin, batch, pooled, NH, H);
    k_readout   <<<blocks((long long)G * T), BT, 0, stream>>>(pooled, counts, Wout, bout,
                                                              out, G, H, T);
}
